// FullSpatialAttention_38276748542168
// MI455X (gfx1250) — compile-verified, matrix operand order checked by reading, not run on hardware
//
#include <hip/hip_runtime.h>

// ---------------- problem constants ----------------
#define BATCH 8
#define CCH   256
#define NHEAD 8
#define SEG   64
#define NTOK  5440          // 64*64 + 32*32 + 16*16 + 8*8
#define MROWS (BATCH * NTOK) // 43520
#define HD    32            // head dim
#define EPS   1e-5f

// ---------------- WMMA vector types ----------------
typedef __attribute__((ext_vector_type(16))) __bf16 v16bf;
typedef __attribute__((ext_vector_type(8)))  __bf16 v8bf;
typedef __attribute__((ext_vector_type(8)))  float  v8f;

// ---------------- fp32 -> bf16 convert ----------------
__global__ void cvt_bf16_kernel(const float* __restrict__ in, __bf16* __restrict__ out, int n) {
    int i = blockIdx.x * blockDim.x + threadIdx.x;
    if (i < n) out[i] = (__bf16)in[i];
}

// ---------------- bf16 WMMA GEMM: Out[M,N] = A[M,K] * W[N,K]^T (+bias) ----------------
// wave tile: 16(M) x 64(N), 4 accumulators, K stepped by 32.
__global__ __launch_bounds__(128)
void gemm_bf16_kernel(const __bf16* __restrict__ A, const __bf16* __restrict__ W,
                      const float* __restrict__ bias, float* __restrict__ Out,
                      int M, int Nout, int K) {
    const int lane = threadIdx.x & 31;
    const int wave = threadIdx.x >> 5;
    const int half = lane >> 4;     // K-half selector (A/B 16-bit layout)
    const int r    = lane & 15;     // A row within tile / B column within subtile
    const int tileM = (blockIdx.x * 4 + wave) * 16;
    const int tileN = blockIdx.y * 64;

    v8f acc[4] = {};

    const __bf16* arow = A + (size_t)(tileM + r) * K;
    const __bf16* wrow = W + (size_t)(tileN + r) * K;

    for (int k0 = 0; k0 < K; k0 += 32) {
        v16bf af;
        ((v8bf*)&af)[0] = *(const v8bf*)(arow + k0 + 8 * half);
        ((v8bf*)&af)[1] = *(const v8bf*)(arow + k0 + 16 + 8 * half);
#pragma unroll
        for (int tt = 0; tt < 4; ++tt) {
            const __bf16* wr = wrow + (size_t)tt * 16 * K;
            v16bf bfr;
            ((v8bf*)&bfr)[0] = *(const v8bf*)(wr + k0 + 8 * half);
            ((v8bf*)&bfr)[1] = *(const v8bf*)(wr + k0 + 16 + 8 * half);
            acc[tt] = __builtin_amdgcn_wmma_f32_16x16x32_bf16(
                false, af, false, bfr, (short)0, acc[tt], false, false);
        }
    }

#pragma unroll
    for (int tt = 0; tt < 4; ++tt) {
        const int ncol = tileN + tt * 16 + r;
        const float bv = bias ? bias[ncol] : 0.0f;
        float* orow = Out + (size_t)(tileM + 8 * half) * Nout + ncol;
#pragma unroll
        for (int j = 0; j < 8; ++j)
            orow[(size_t)j * Nout] = acc[tt][j] + bv;
    }
}

// ---------------- depthwise conv (per scale i, per segment s) ----------------
// reads k/v slice directly from qkv buffer; writes dwout[b2][p][c] (c-inner)
__global__ void dwconv_kernel(const float* __restrict__ qkvbuf, const float* __restrict__ dw,
                              float* __restrict__ out, int H, int W, int off, int cbase,
                              int ks, int pad) {
    const int HW = H * W;
    int idx = blockIdx.x * blockDim.x + threadIdx.x;
    if (idx >= 16 * 64 * HW) return;
    const int c    = idx & 63;
    const int rest = idx >> 6;
    const int p    = rest % HW;
    const int b2   = rest / HW;
    const int h = p / W, w = p % W;
    const int kv = b2 >> 3, b = b2 & 7;
    const float* wk = dw + c * ks * ks;
    const size_t base = (size_t)(b * NTOK + off) * 768 + 256 + kv * 256 + cbase + c;
    float acc = 0.0f;
    for (int dy = 0; dy < ks; ++dy) {
        const int hy = h + dy - pad;
        if ((unsigned)hy >= (unsigned)H) continue;
        for (int dx = 0; dx < ks; ++dx) {
            const int wx = w + dx - pad;
            if ((unsigned)wx >= (unsigned)W) continue;
            acc += wk[dy * ks + dx] * qkvbuf[base + (size_t)(hy * W + wx) * 768];
        }
    }
    out[((size_t)b2 * HW + p) * 64 + c] = acc;
}

// ---------------- pointwise 64x64 conv ----------------
// in: dwout[b2][p][c] -> out: pwout[b2][o][p]
__global__ void pwconv_kernel(const float* __restrict__ t_in, const float* __restrict__ pw,
                              float* __restrict__ out, int HW) {
    int idx = blockIdx.x * blockDim.x + threadIdx.x;
    if (idx >= 16 * 64 * HW) return;
    const int p  = idx % HW;
    const int o  = (idx / HW) & 63;
    const int b2 = idx / (HW * 64);
    const float* trow = t_in + ((size_t)b2 * HW + p) * 64;
    const float* wrow = pw + o * 64;
    float a = 0.0f;
#pragma unroll
    for (int c = 0; c < 64; ++c) a += trow[c] * wrow[c];
    out[((size_t)b2 * 64 + o) * HW + p] = a;
}

// ---------------- BatchNorm batch-stats: one block per channel ----------------
__global__ void bnstats_kernel(const float* __restrict__ src, const float* __restrict__ qkvbuf,
                               int cbase, int off, int HW, float* __restrict__ stats) {
    const int c = blockIdx.x;
    const int t = threadIdx.x;
    const int total = 16 * HW;
    float s1 = 0.0f, s2 = 0.0f;
    for (int e = t; e < total; e += 256) {
        const int b2 = e / HW, p = e % HW;
        float v;
        if (src) {
            v = src[((size_t)b2 * 64 + c) * HW + p];
        } else {
            const int kv = b2 >> 3, b = b2 & 7;
            v = qkvbuf[((size_t)(b * NTOK + off + p)) * 768 + 256 + kv * 256 + cbase + c];
        }
        s1 += v; s2 += v * v;
    }
    __shared__ float r1[256], r2[256];
    r1[t] = s1; r2[t] = s2;
    __syncthreads();
    for (int st = 128; st > 0; st >>= 1) {
        if (t < st) { r1[t] += r1[t + st]; r2[t] += r2[t + st]; }
        __syncthreads();
    }
    if (t == 0) {
        const float inv = 1.0f / (float)total;
        const float m   = r1[0] * inv;
        const float var = r2[0] * inv - m * m;
        stats[c]      = m;
        stats[64 + c] = rsqrtf(var + EPS);
    }
}

// ---------------- BN apply + ReLU, scatter into kv_agg [2,B,NH,32,N] ----------------
__global__ void bnapply_kernel(const float* __restrict__ src, const float* __restrict__ qkvbuf,
                               int cbase, int off, int HW, const float* __restrict__ stats,
                               const float* __restrict__ gamma, const float* __restrict__ beta,
                               float* __restrict__ kvagg, int s) {
    int idx = blockIdx.x * blockDim.x + threadIdx.x;
    if (idx >= 16 * 64 * HW) return;
    const int p  = idx % HW;
    const int c  = (idx / HW) & 63;
    const int b2 = idx / (HW * 64);
    const int kv = b2 >> 3, b = b2 & 7;
    float v;
    if (src) {
        v = src[((size_t)b2 * 64 + c) * HW + p];
    } else {
        v = qkvbuf[((size_t)(b * NTOK + off + p)) * 768 + 256 + kv * 256 + cbase + c];
    }
    const float xh = (v - stats[c]) * stats[64 + c];
    float y = xh * gamma[c] + beta[c];
    y = fmaxf(y, 0.0f);
    const int nh = c >> 3;
    const int hd = s * 8 + (c & 7);
    kvagg[((size_t)(((kv * 8 + b) * 8 + nh) * 32) + hd) * NTOK + off + p] = y;
}

// ---------------- K-softmax stats (max & sum over tokens) per (b,h,hd) ----------------
__global__ void ksm_stats_kernel(const float* __restrict__ kvagg, float* __restrict__ kstats) {
    const int bh = blockIdx.x;   // b*8+nh
    const int t  = threadIdx.x;
    const int hd = t >> 3, sub = t & 7;
    const float* row = kvagg + ((size_t)bh * 32 + hd) * NTOK;
    __shared__ float red[256];
    __shared__ float smax[32];
    float m = -1e30f;
    for (int n = sub; n < NTOK; n += 8) m = fmaxf(m, row[n]);
    red[t] = m;
    __syncthreads();
    if (sub == 0) {
        float mm = red[t];
        for (int j = 1; j < 8; ++j) mm = fmaxf(mm, red[t + j]);
        smax[hd] = mm;
    }
    __syncthreads();
    const float mv = smax[hd];
    float ssum = 0.0f;
    for (int n = sub; n < NTOK; n += 8) ssum += expf(row[n] - mv);
    red[t] = ssum;
    __syncthreads();
    if (sub == 0) {
        float ss = red[t];
        for (int j = 1; j < 8; ++j) ss += red[t + j];
        kstats[bh * 64 + hd]      = mv;
        kstats[bh * 64 + 32 + hd] = ss;
    }
}

// ---------------- ktv[b,h,kd,vd] = sum_n softmax(k)[n,kd] * v[n,vd] ----------------
__global__ void ktv_kernel(const float* __restrict__ kvagg, const float* __restrict__ kstats,
                           float* __restrict__ ktv) {
    const int bh = blockIdx.x;
    const int t  = threadIdx.x;
    __shared__ float ks[64 * 33];
    __shared__ float vs[64 * 33];
    __shared__ float smax[32];
    __shared__ float srcp[32];
    if (t < 32) {
        smax[t] = kstats[bh * 64 + t];
        srcp[t] = 1.0f / kstats[bh * 64 + 32 + t];
    }
    const float* kb = kvagg + (size_t)bh * 32 * NTOK;          // kv=0
    const float* vb = kvagg + (size_t)(64 + bh) * 32 * NTOK;   // kv=1
    const int kd = t & 31;
    const int vg = (t >> 5) * 4;
    float a0 = 0, a1 = 0, a2 = 0, a3 = 0;
    for (int n0 = 0; n0 < NTOK; n0 += 64) {
        __syncthreads();
#pragma unroll
        for (int j = 0; j < 8; ++j) {
            const int e  = t + j * 256;
            const int hd = e >> 6, jj = e & 63;
            ks[jj * 33 + hd] = expf(kb[(size_t)hd * NTOK + n0 + jj] - smax[hd]) * srcp[hd];
            vs[jj * 33 + hd] = vb[(size_t)hd * NTOK + n0 + jj];
        }
        __syncthreads();
#pragma unroll 8
        for (int j = 0; j < 64; ++j) {
            const float kk = ks[j * 33 + kd];
            const float* vr = &vs[j * 33 + vg];
            a0 += kk * vr[0]; a1 += kk * vr[1]; a2 += kk * vr[2]; a3 += kk * vr[3];
        }
    }
    float* o = ktv + ((size_t)bh * 32 + kd) * 32 + vg;
    o[0] = a0; o[1] = a1; o[2] = a2; o[3] = a3;
}

// ---------------- Q channel-softmax + eff = q_sm @ ktv, store bf16 ----------------
__global__ void qeff_kernel(const float* __restrict__ qkvbuf, const float* __restrict__ ktv,
                            __bf16* __restrict__ eff) {
    const int bh = blockIdx.x;
    const int b = bh >> 3, nh = bh & 7;
    const int t = threadIdx.x;
    __shared__ float lk[1024];
    for (int j = t; j < 1024; j += 256) lk[j] = ktv[(size_t)bh * 1024 + j];
    __syncthreads();
    const int n = blockIdx.y * 256 + t;
    if (n >= NTOK) return;
    const float* q = qkvbuf + (size_t)(b * NTOK + n) * 768 + nh * 32;
    float qv[32];
    float m = -1e30f;
#pragma unroll
    for (int k = 0; k < 32; ++k) { qv[k] = q[k]; m = fmaxf(m, qv[k]); }
    float ssum = 0.0f;
#pragma unroll
    for (int k = 0; k < 32; ++k) { qv[k] = expf(qv[k] - m); ssum += qv[k]; }
    const float r = 1.0f / ssum;
    __bf16* o = eff + (size_t)(b * NTOK + n) * 256 + nh * 32;
#pragma unroll
    for (int vd = 0; vd < 32; ++vd) {
        float a = 0.0f;
#pragma unroll
        for (int k = 0; k < 32; ++k) a += qv[k] * lk[k * 32 + vd];
        o[vd] = (__bf16)(a * r);
    }
}

// ==================================================================
extern "C" void kernel_launch(void* const* d_in, const int* in_sizes, int n_in,
                              void* d_out, int out_size, void* d_ws, size_t ws_size,
                              hipStream_t stream) {
    const float* x      = (const float*)d_in[0];
    const float* w_qkv  = (const float*)d_in[1];
    const float* w_proj = (const float*)d_in[2];
    const float* b_proj = (const float*)d_in[3];
    const float* dw3    = (const float*)d_in[4];
    const float* pw3    = (const float*)d_in[5];
    const float* dw5    = (const float*)d_in[6];
    const float* pw5    = (const float*)d_in[7];
    const float* dw7    = (const float*)d_in[8];
    const float* pw7    = (const float*)d_in[9];
    const float* bn_g   = (const float*)d_in[10];
    const float* bn_b   = (const float*)d_in[11];
    float* out = (float*)d_out;

    // ---- workspace carve-up (256B aligned) ----
    char* w = (char*)d_ws;
    auto alloc = [&](size_t bytes) -> char* {
        char* p = w;
        w += (bytes + 255) & ~(size_t)255;
        return p;
    };
    float*  qkvbuf = (float*)alloc((size_t)MROWS * 768 * 4);          // 133.7 MB
    float*  kvagg  = (float*)alloc((size_t)2 * 8 * 8 * 32 * NTOK * 4); // 89.1 MB
    __bf16* xb     = (__bf16*)alloc((size_t)MROWS * 256 * 2);          // 22.3 MB (reused as eff)
    __bf16* wqkvb  = (__bf16*)alloc((size_t)768 * 256 * 2);
    __bf16* wprojb = (__bf16*)alloc((size_t)256 * 256 * 2);
    float*  dwout  = (float*)alloc((size_t)16 * 64 * 4096 * 4);        // 16.8 MB
    float*  pwout  = (float*)alloc((size_t)16 * 64 * 4096 * 4);        // 16.8 MB
    float*  bnst   = (float*)alloc(128 * 4);
    float*  kstats = (float*)alloc(64 * 64 * 4);
    float*  ktvbuf = (float*)alloc((size_t)64 * 1024 * 4);
    __bf16* effb   = xb;  // x is dead after the qkv GEMM

    // ---- 1) fp32 -> bf16 conversions ----
    {
        int n = MROWS * 256;
        cvt_bf16_kernel<<<(n + 255) / 256, 256, 0, stream>>>(x, xb, n);
        n = 768 * 256;
        cvt_bf16_kernel<<<(n + 255) / 256, 256, 0, stream>>>(w_qkv, wqkvb, n);
        n = 256 * 256;
        cvt_bf16_kernel<<<(n + 255) / 256, 256, 0, stream>>>(w_proj, wprojb, n);
    }

    // ---- 2) qkv GEMM (WMMA bf16, fp32 acc): [43520,768] ----
    {
        dim3 g(MROWS / 64, 768 / 64);
        gemm_bf16_kernel<<<g, 128, 0, stream>>>(xb, wqkvb, nullptr, qkvbuf, MROWS, 768, 256);
    }

    // ---- 3) aggregators: 4 spatial scales x 4 channel segments ----
    const int    Hs[4]  = {64, 32, 16, 8};
    const int    Ws[4]  = {64, 32, 16, 8};
    const int    offs[4] = {0, 4096, 5120, 5376};
    const float* dws[3] = {dw3, dw5, dw7};
    const float* pws[3] = {pw3, pw5, pw7};
    const int    kss[3] = {3, 5, 7};

    for (int i = 0; i < 4; ++i) {
        const int H = Hs[i], W = Ws[i], off = offs[i];
        const int HW = H * W;
        const int total = 16 * 64 * HW;
        const int nb = (total + 255) / 256;
        for (int s = 0; s < 4; ++s) {
            const float* src = nullptr;
            if (s > 0) {
                const int k = kss[s - 1];
                dwconv_kernel<<<nb, 256, 0, stream>>>(qkvbuf, dws[s - 1] + i * 64 * k * k,
                                                      dwout, H, W, off, 64 * s, k, k / 2);
                pwconv_kernel<<<nb, 256, 0, stream>>>(dwout, pws[s - 1] + i * 64 * 64,
                                                      pwout, HW);
                src = pwout;
            }
            bnstats_kernel<<<64, 256, 0, stream>>>(src, qkvbuf, 64 * s, off, HW, bnst);
            bnapply_kernel<<<nb, 256, 0, stream>>>(src, qkvbuf, 64 * s, off, HW, bnst,
                                                   bn_g + (i * 4 + s) * 64,
                                                   bn_b + (i * 4 + s) * 64, kvagg, s);
        }
    }

    // ---- 4) K token-softmax stats + ktv = softmax(k)^T v ----
    ksm_stats_kernel<<<64, 256, 0, stream>>>(kvagg, kstats);
    ktv_kernel<<<64, 256, 0, stream>>>(kvagg, kstats, ktvbuf);

    // ---- 5) Q channel-softmax + eff (bf16) ----
    {
        dim3 g(64, (NTOK + 255) / 256);
        qeff_kernel<<<g, 256, 0, stream>>>(qkvbuf, ktvbuf, effb);
    }

    // ---- 6) output projection GEMM (WMMA) + bias ----
    {
        dim3 g(MROWS / 64, 256 / 64);
        gemm_bf16_kernel<<<g, 128, 0, stream>>>(effb, wprojb, b_proj, out, MROWS, 256, 256);
    }
}
